// SpatialTemporalAttention_59201829208591
// MI455X (gfx1250) — compile-verified, matrix operand order checked by reading, not run on hardware
//
#include <hip/hip_runtime.h>
#include <hip/hip_bf16.h>
#include <cfloat>

typedef __attribute__((ext_vector_type(16))) _Float16 v16h;
typedef __attribute__((ext_vector_type(8)))  _Float16 v8h;
typedef __attribute__((ext_vector_type(8)))  float    v8f;

#define B_ 8
#define T_ 20
#define N_ 512
#define D_ 64
#define AS 520   // s_alpha row stride in halves (padded: banks 4*row mod 64)
#define HS 264   // s_hT row stride in halves (padded, 16B-aligned rows)

// ---------------- Kernel 1: masked mean over nodes -> h_mean (B,T,D) ----------
__global__ void k_hmean(const float* __restrict__ h,
                        const unsigned char* __restrict__ mask,
                        float* __restrict__ hmean) {
  int bt = blockIdx.x;            // 0..159
  int d  = threadIdx.x;           // 0..63
  const float* hp = h + (size_t)bt * N_ * D_;
  const unsigned char* mp = mask + (size_t)bt * N_;
  float acc = 0.f, cnt = 0.f;
  for (int n = 0; n < N_; ++n) {
    float m = mp[n] ? 1.f : 0.f;
    acc += m * hp[n * D_ + d];
    cnt += m;
  }
  hmean[bt * D_ + d] = acc / fmaxf(cnt, 1.f);
}

// ---------------- Kernel 2: m_b = hm hm^T, beta = softmax rows (per batch) ----
__global__ void k_beta(const float* __restrict__ hmean,
                       float* __restrict__ beta) {
  int b = blockIdx.x;
  __shared__ float s_hm[T_ * D_];
  __shared__ float s_m[T_ * T_];
  for (int e = threadIdx.x; e < T_ * D_; e += blockDim.x)
    s_hm[e] = hmean[b * T_ * D_ + e];
  __syncthreads();
  for (int e = threadIdx.x; e < T_ * T_; e += blockDim.x) {
    int t = e / T_, u = e % T_;
    float acc = 0.f;
    #pragma unroll
    for (int d = 0; d < D_; ++d) acc += s_hm[t * D_ + d] * s_hm[u * D_ + d];
    s_m[e] = acc;
  }
  __syncthreads();
  if (threadIdx.x < T_) {
    int t = threadIdx.x;
    float mx = -FLT_MAX;
    #pragma unroll
    for (int u = 0; u < T_; ++u) mx = fmaxf(mx, s_m[t * T_ + u]);
    float ex[T_]; float sum = 0.f;
    #pragma unroll
    for (int u = 0; u < T_; ++u) { ex[u] = __expf(s_m[t * T_ + u] - mx); sum += ex[u]; }
    float inv = 1.f / sum;
    #pragma unroll
    for (int u = 0; u < T_; ++u) beta[(b * T_ + t) * T_ + u] = ex[u] * inv;
  }
}

// ---------------- Kernel 3: fused spatial attention, WMMA f16 -----------------
// One block = one (b,t) and one 16-row query block. 128 threads = 4 waves.
// Wave w softmaxes rows w*4..w*4+3 and owns output D-tile d = w*16..w*16+15.
__global__ void __launch_bounds__(128)
k_spatial(const float* __restrict__ h,
          const float* __restrict__ vel,
          const float* __restrict__ adj,
          const unsigned char* __restrict__ mask,
          float* __restrict__ out) {
  const int blk  = blockIdx.x;
  const int ib   = blk & 31;        // row-block within (b,t)
  const int bt   = blk >> 5;        // 0..159
  const int tid  = threadIdx.x;
  const int lane = tid & 31;
  const int wv   = tid >> 5;

  __shared__ __align__(16) float    s_vel[N_ * 2];
  __shared__             float      s_mask[N_];
  __shared__ __align__(16) _Float16 s_alpha[16 * AS];
  __shared__ __align__(16) _Float16 s_hT[D_ * HS];

  const float* velp = vel + (size_t)bt * N_ * 2;
  const unsigned char* mp = mask + (size_t)bt * N_;
  const float* hp = h + (size_t)bt * N_ * D_;

  for (int e = tid; e < N_ * 2; e += 128) s_vel[e]  = velp[e];
  for (int e = tid; e < N_;     e += 128) s_mask[e] = mp[e] ? 1.f : 0.f;
  __syncthreads();

  // ---- logits (registers) + row softmax: 8 lanes per row, 64 elems per lane
  {
    const int rloc = wv * 4 + (lane >> 3);
    const int sub  = lane & 7;
    const int ig   = ib * 16 + rloc;
    const float mi  = s_mask[ig];
    const float vix = s_vel[2 * ig], viy = s_vel[2 * ig + 1];
    const float* arow = adj + ((size_t)bt * N_ + ig) * N_;
    float lg[64];
    #pragma unroll
    for (int jj = 0; jj < 64; ++jj) {
      int j = sub + jj * 8;
      float dx = vix - s_vel[2 * j];
      float dy = viy - s_vel[2 * j + 1];
      float v  = arow[j] - (dx * dx + dy * dy);   // GAMMA=1, LAMBDA=1
      bool ok  = (mi != 0.f) && (s_mask[j] != 0.f);
      lg[jj] = ok ? v : -3.402823466e38f;
    }
    float mx = -FLT_MAX;
    #pragma unroll
    for (int jj = 0; jj < 64; ++jj) mx = fmaxf(mx, lg[jj]);
    mx = fmaxf(mx, __shfl_xor(mx, 1));
    mx = fmaxf(mx, __shfl_xor(mx, 2));
    mx = fmaxf(mx, __shfl_xor(mx, 4));
    float sum = 0.f;
    #pragma unroll
    for (int jj = 0; jj < 64; ++jj) { lg[jj] = __expf(lg[jj] - mx); sum += lg[jj]; }
    sum += __shfl_xor(sum, 1);
    sum += __shfl_xor(sum, 2);
    sum += __shfl_xor(sum, 4);
    float scale = (mi != 0.f) ? (1.f / sum) : 0.f;   // invalid rows -> alpha 0
    #pragma unroll
    for (int jj = 0; jj < 64; ++jj)
      s_alpha[rloc * AS + sub + jj * 8] = (_Float16)(lg[jj] * scale);
  }
  __syncthreads();

  // ---- alpha(16x512,f16) @ h(512x64,f16) via v_wmma_f32_16x16x32_f16
  const int col = lane & 15;
  const int kb  = (lane & 16) ? 8 : 0;
  v8f acc = {};
  for (int p = 0; p < 2; ++p) {
    // stage h[j in [p*256, p*256+256)][d] -> s_hT[d][j-local], f16 transposed
    #pragma unroll 4
    for (int it = 0; it < 32; ++it) {
      int e  = tid + it * 128;          // 4096 float4 groups
      int jl = e >> 4;                  // 0..255
      int dq = e & 15;
      const float4 hv = ((const float4*)hp)[((size_t)(p * 256 + jl)) * 16 + dq];
      int d0 = dq * 4;
      s_hT[(d0 + 0) * HS + jl] = (_Float16)hv.x;
      s_hT[(d0 + 1) * HS + jl] = (_Float16)hv.y;
      s_hT[(d0 + 2) * HS + jl] = (_Float16)hv.z;
      s_hT[(d0 + 3) * HS + jl] = (_Float16)hv.w;
    }
    __syncthreads();
    const _Float16* abase = &s_alpha[col * AS + p * 256 + kb];
    const _Float16* bbase = &s_hT[(wv * 16 + col) * HS + kb];
    #pragma unroll
    for (int k0 = 0; k0 < 256; k0 += 32) {
      v8h alo = *(const v8h*)(abase + k0);
      v8h ahi = *(const v8h*)(abase + k0 + 16);
      v8h blo = *(const v8h*)(bbase + k0);
      v8h bhi = *(const v8h*)(bbase + k0 + 16);
      v16h av, bv;
      #pragma unroll
      for (int i = 0; i < 8; ++i) {
        av[i] = alo[i]; av[i + 8] = ahi[i];
        bv[i] = blo[i]; bv[i + 8] = bhi[i];
      }
      acc = __builtin_amdgcn_wmma_f32_16x16x32_f16(
          false, av, false, bv, (short)0, acc, false, false);
    }
    __syncthreads();
  }

  // ---- epilogue: out = h + h_spatial (mask applied in final kernel)
  const int mbase = (lane & 16) ? 8 : 0;
  #pragma unroll
  for (int v = 0; v < 8; ++v) {
    int i = ib * 16 + mbase + v;
    int d = wv * 16 + col;
    size_t idx = ((size_t)bt * N_ + i) * D_ + d;
    out[idx] = hp[(size_t)i * D_ + d] + acc[v];
  }
}

// ---------------- Kernel 4: add temporal mix, apply mask, finalize ------------
__global__ void __launch_bounds__(256)
k_temporal(const float* __restrict__ h,
           const unsigned char* __restrict__ mask,
           const float* __restrict__ beta,
           float* __restrict__ out) {
  int b = blockIdx.x >> 7;
  int c = blockIdx.x & 127;
  int f = c * 256 + threadIdx.x;      // 0..32767 over (n,d)
  int n = f >> 6;
  __shared__ float s_beta[T_ * T_];
  for (int e = threadIdx.x; e < T_ * T_; e += 256) s_beta[e] = beta[b * T_ * T_ + e];
  __syncthreads();
  float hv[T_];
  #pragma unroll
  for (int u = 0; u < T_; ++u)
    hv[u] = h[((size_t)(b * T_ + u)) * (N_ * D_) + f];
  #pragma unroll
  for (int t = 0; t < T_; ++t) {
    float acc = 0.f;
    #pragma unroll
    for (int u = 0; u < T_; ++u) acc += s_beta[t * T_ + u] * hv[u];
    size_t idx = ((size_t)(b * T_ + t)) * (N_ * D_) + f;
    float s = out[idx];                      // h + h_spatial
    float m = mask[(size_t)(b * T_ + t) * N_ + n] ? 1.f : 0.f;
    out[idx] = (s + acc) * m;
  }
}

extern "C" void kernel_launch(void* const* d_in, const int* in_sizes, int n_in,
                              void* d_out, int out_size, void* d_ws, size_t ws_size,
                              hipStream_t stream) {
  const float* h   = (const float*)d_in[0];
  const float* vel = (const float*)d_in[1];
  const float* adj = (const float*)d_in[2];
  const unsigned char* mask = (const unsigned char*)d_in[3];
  float* out = (float*)d_out;

  float* hmean = (float*)d_ws;              // B*T*D floats
  float* beta  = hmean + B_ * T_ * D_;      // B*T*T floats

  k_hmean  <<<B_ * T_,            D_,  0, stream>>>(h, mask, hmean);
  k_beta   <<<B_,                 128, 0, stream>>>(hmean, beta);
  k_spatial<<<B_ * T_ * (N_ / 16),128, 0, stream>>>(h, vel, adj, mask, out);
  k_temporal<<<B_ * 128,          256, 0, stream>>>(h, mask, beta, out);
}
